// IdentityDecouplingModule_16312285790881
// MI455X (gfx1250) — matrix operand order repaired and checked by reading, not compiled, not run on hardware
//
#include <hip/hip_runtime.h>
#include <hip/hip_bf16.h>

// ---------------------------------------------------------------------------
// Problem constants (reference: B=1, C=1280, H=W=64 -> S=4096, HID=128)
// ---------------------------------------------------------------------------
#define S_LEN 4096
#define C_DIM 1280
#define HID   128
#define D_ID  1536
#define EPS   1e-5f

typedef __attribute__((ext_vector_type(16))) __bf16 v16bf;
typedef __attribute__((ext_vector_type(8)))  float  v8f;

union Frag {            // 32 bytes: one WMMA bf16 operand, filled via 2x b128 LDS loads
    v16bf v;
    uint4 q[2];
};

// gfx1250 async global->LDS copy (ASYNCcnt-tracked, no VGPR round trip)
#if defined(__has_builtin)
#if __has_builtin(__builtin_amdgcn_global_load_async_to_lds_b128)
#define USE_ASYNC_LDS 1
#endif
#endif

#if defined(USE_ASYNC_LDS)
// Builtin parameter type (from hipcc diagnostic): int __vector_size__(16) *
typedef int b128vec __attribute__((__vector_size__(4 * sizeof(int))));

__device__ __forceinline__ void async_copy16(const void* gsrc, void* ldst)
{
    __builtin_amdgcn_global_load_async_to_lds_b128(
        (__attribute__((address_space(1))) b128vec*)(void*)gsrc,
        (__attribute__((address_space(3))) b128vec*)ldst,
        0, 0);
}
#endif

// ---------------------------------------------------------------------------
// Generic bf16 WMMA GEMM:  out[M,N] = (A[M,K] * B[N,K]^T * scale + bias) * mask
//   - A, B row-major bf16;  B stored as (N rows, K cols) == PyTorch weight layout
//   - output either bf16 (outb) or fp32 (outf); optional transposed store
// One wave computes a 16(M) x 64(N) tile; K stepped by 32; LDS double-buffered
// with async global->LDS copies. Fragment layouts per CDNA5 ISA 7.12.2 (wave32).
// ---------------------------------------------------------------------------
__global__ __launch_bounds__(32)
void gemm_bf16_wmma(const __hip_bfloat16* __restrict__ A,
                    const __hip_bfloat16* __restrict__ B,
                    const float* __restrict__ bias,
                    const float* __restrict__ maskv,
                    __hip_bfloat16* __restrict__ outb,
                    float* __restrict__ outf,
                    int M, int N, int K,
                    int lda, int ldb, int ldc,
                    float scale, int store_transposed)
{
    __shared__ unsigned short lds_a[2][16 * 32];   // 16 rows x 32 K, bf16, x2 buffers
    __shared__ unsigned short lds_b[2][64 * 32];   // 64 rows x 32 K, bf16, x2 buffers

    const int lane = threadIdx.x;
    const int m0   = blockIdx.x * 16;
    const int n0   = blockIdx.y * 64;
    const int half = lane >> 4;      // K-half owned by this lane
    const int r    = lane & 15;      // row (A) / column (B) index

    // per-lane 16B-chunk coordinates (fixed trip counts -> no exec-mask branches)
    const int ar0 = lane >> 2;             // A chunk 0: rows 0..7
    const int ar1 = (lane >> 2) + 8;       // A chunk 1: rows 8..15
    const int ac  = (lane & 3) * 8;        // chunk column (bf16 elements)
    const int br  = lane >> 2;             // B chunk base row
    const int bc  = (lane & 3) * 8;

    // Stage one 16x32 A tile + 64x32 B tile for K-offset kk into buffer `buf`.
    auto issue_tile = [&](int kk, int buf) {
#if defined(USE_ASYNC_LDS)
        async_copy16(A + (size_t)(m0 + ar0) * lda + kk + ac, &lds_a[buf][ar0 * 32 + ac]);
        async_copy16(A + (size_t)(m0 + ar1) * lda + kk + ac, &lds_a[buf][ar1 * 32 + ac]);
        #pragma unroll
        for (int i = 0; i < 8; ++i) {
            const int row = br + i * 8;
            async_copy16(B + (size_t)(n0 + row) * ldb + kk + bc, &lds_b[buf][row * 32 + bc]);
        }
#else
        uint4 t[10];
        t[0] = *reinterpret_cast<const uint4*>(A + (size_t)(m0 + ar0) * lda + kk + ac);
        t[1] = *reinterpret_cast<const uint4*>(A + (size_t)(m0 + ar1) * lda + kk + ac);
        #pragma unroll
        for (int i = 0; i < 8; ++i)
            t[2 + i] = *reinterpret_cast<const uint4*>(B + (size_t)(n0 + br + i * 8) * ldb + kk + bc);
        *reinterpret_cast<uint4*>(&lds_a[buf][ar0 * 32 + ac]) = t[0];
        *reinterpret_cast<uint4*>(&lds_a[buf][ar1 * 32 + ac]) = t[1];
        #pragma unroll
        for (int i = 0; i < 8; ++i)
            *reinterpret_cast<uint4*>(&lds_b[buf][(br + i * 8) * 32 + bc]) = t[2 + i];
#endif
    };

    v8f acc[4] = {};                 // four 16x16 fp32 accumulators along N

    issue_tile(0, 0);
    int buf = 0;
    for (int kk = 0; kk < K; kk += 32) {
        const bool more = (kk + 32) < K;
        if (more) issue_tile(kk + 32, buf ^ 1);     // prefetch next tile (other buffer)

#if defined(USE_ASYNC_LDS)
        // Async loads complete in order: with 20 outstanding, waiting to <=10
        // guarantees the current tile's 10 copies have landed in LDS.
        if (more) asm volatile("s_wait_asynccnt 0xa" ::: "memory");
        else      asm volatile("s_wait_asynccnt 0x0" ::: "memory");
#else
        __syncthreads();
#endif

        // A fragment: row r, K = {half*8 .. +7} and {16+half*8 .. +7}  (2x b128)
        Frag fa;
        fa.q[0] = *reinterpret_cast<const uint4*>(&lds_a[buf][r * 32 + half * 8]);
        fa.q[1] = *reinterpret_cast<const uint4*>(&lds_a[buf][r * 32 + 16 + half * 8]);

        #pragma unroll
        for (int t = 0; t < 4; ++t) {
            // B fragment: column (t*16 + r), K = {half*8 .. +7} and {16+half*8 .. +7} (2x b128)
            Frag fb;
            fb.q[0] = *reinterpret_cast<const uint4*>(&lds_b[buf][(t * 16 + r) * 32 + half * 8]);
            fb.q[1] = *reinterpret_cast<const uint4*>(&lds_b[buf][(t * 16 + r) * 32 + 16 + half * 8]);
            acc[t] = __builtin_amdgcn_wmma_f32_16x16x32_bf16(
                false, fa.v, false, fb.v, (short)0, acc[t], false, false);
        }

#if defined(USE_ASYNC_LDS)
        // Fragment ds_loads must retire before this buffer is overwritten next iter.
        asm volatile("s_wait_dscnt 0x0" ::: "memory");
#else
        __syncthreads();
#endif
        buf ^= 1;
    }

    // --- epilogue: lane L, vgpr v  ->  (m = half*8 + v, n = r) in each subtile ---
    #pragma unroll
    for (int t = 0; t < 4; ++t) {
        const int n = n0 + t * 16 + r;
        const float b  = bias  ? bias[n]  : 0.0f;
        const float mk = maskv ? maskv[n] : 1.0f;
        #pragma unroll
        for (int v = 0; v < 8; ++v) {
            const int m = m0 + half * 8 + v;
            const float val = (acc[t][v] * scale + b) * mk;
            const size_t idx = store_transposed ? (size_t)n * ldc + m
                                                : (size_t)m * ldc + n;
            if (outb) outb[idx] = __float2bfloat16(val);
            else      outf[idx] = val;
        }
    }
}

// ---------------------------------------------------------------------------
// fp32 -> bf16 weight conversion
// ---------------------------------------------------------------------------
__global__ void f32_to_bf16_kernel(const float* __restrict__ src,
                                   __hip_bfloat16* __restrict__ dst, int n)
{
    int i = blockIdx.x * blockDim.x + threadIdx.x;
    if (i < n) dst[i] = __float2bfloat16(src[i]);
}

// ---------------------------------------------------------------------------
// Per-token LayerNorm over C with transpose: z (C, S) fp32 -> zf (S, C) bf16.
// Block = (32 tokens, 8 channel-strips) so global reads stay coalesced in x.
// ---------------------------------------------------------------------------
__global__ __launch_bounds__(256)
void ln_rows_kernel(const float* __restrict__ z, __hip_bfloat16* __restrict__ zf)
{
    __shared__ float s_sum[8][32];
    __shared__ float s_sq[8][32];
    const int tx = threadIdx.x, ty = threadIdx.y;
    const int s = blockIdx.x * 32 + tx;

    float sum = 0.f, sq = 0.f;
    for (int c = ty; c < C_DIM; c += 8) {
        float v = z[(size_t)c * S_LEN + s];
        sum += v; sq += v * v;
    }
    s_sum[ty][tx] = sum; s_sq[ty][tx] = sq;
    __syncthreads();
    if (ty == 0) {
        float a = 0.f, b = 0.f;
        #pragma unroll
        for (int j = 0; j < 8; ++j) { a += s_sum[j][tx]; b += s_sq[j][tx]; }
        const float mean = a / C_DIM;
        const float var  = b / C_DIM - mean * mean;
        s_sum[0][tx] = mean;
        s_sq[0][tx]  = rsqrtf(var + EPS);
    }
    __syncthreads();
    const float mean = s_sum[0][tx], rstd = s_sq[0][tx];
    for (int c = ty; c < C_DIM; c += 8) {
        float v = z[(size_t)c * S_LEN + s];
        zf[(size_t)s * C_DIM + c] = __float2bfloat16((v - mean) * rstd);
    }
}

// ---------------------------------------------------------------------------
// Identity-feature projections: iv = mlp(cid) (128), cidp = proj(cid) (1280).
// ---------------------------------------------------------------------------
__global__ void small_linear_kernel(const float* __restrict__ cid,
                                    const float* __restrict__ mlp_w,
                                    const float* __restrict__ mlp_b,
                                    const float* __restrict__ proj_w,
                                    const float* __restrict__ proj_b,
                                    float* __restrict__ iv,
                                    float* __restrict__ cidp)
{
    int i = blockIdx.x * blockDim.x + threadIdx.x;
    if (i < HID) {
        float s = mlp_b[i];
        const float* w = mlp_w + (size_t)i * D_ID;
        for (int j = 0; j < D_ID; ++j) s += w[j] * cid[j];
        iv[i] = s;
    } else if (i < HID + C_DIM) {
        int o = i - HID;
        float s = proj_b[o];
        const float* w = proj_w + (size_t)o * D_ID;
        for (int j = 0; j < D_ID; ++j) s += w[j] * cid[j];
        cidp[o] = s;
    }
}

// ---------------------------------------------------------------------------
// Column mask: mask[k] = sigmoid( iv . K_k )
// ---------------------------------------------------------------------------
__global__ void mask_kernel(const __hip_bfloat16* __restrict__ Kb,
                            const float* __restrict__ iv,
                            float* __restrict__ maskv)
{
    int k = blockIdx.x * blockDim.x + threadIdx.x;
    if (k >= S_LEN) return;
    float s = 0.f;
    const __hip_bfloat16* row = Kb + (size_t)k * HID;
    for (int h = 0; h < HID; ++h) s += __bfloat162float(row[h]) * iv[h];
    maskv[k] = 1.f / (1.f + expf(-s));
}

// ---------------------------------------------------------------------------
// In-place row softmax over bf16 logits (one block per row).
// ---------------------------------------------------------------------------
__global__ __launch_bounds__(256)
void softmax_kernel(__hip_bfloat16* __restrict__ P)
{
    __shared__ float red[256];
    const int t = threadIdx.x;
    __hip_bfloat16* p = P + (size_t)blockIdx.x * S_LEN;

    float mx = -3.0e38f;
    for (int k = t; k < S_LEN; k += 256) mx = fmaxf(mx, __bfloat162float(p[k]));
    red[t] = mx; __syncthreads();
    for (int o = 128; o > 0; o >>= 1) { if (t < o) red[t] = fmaxf(red[t], red[t + o]); __syncthreads(); }
    mx = red[0]; __syncthreads();

    float sum = 0.f;
    for (int k = t; k < S_LEN; k += 256) sum += expf(__bfloat162float(p[k]) - mx);
    red[t] = sum; __syncthreads();
    for (int o = 128; o > 0; o >>= 1) { if (t < o) red[t] += red[t + o]; __syncthreads(); }
    const float inv = 1.f / red[0];

    for (int k = t; k < S_LEN; k += 256)
        p[k] = __float2bfloat16(expf(__bfloat162float(p[k]) - mx) * inv);
}

// ---------------------------------------------------------------------------
// Decoupling epilogue (one block per token s):
//   sim = att_s . cidp ; w = sigmoid(sim) ; af = att_s - w*cidp
// Writes af transposed into d_out (C, S) and per-token partial sums for the
// global layer-norm (deterministic two-stage reduction, no atomics).
// ---------------------------------------------------------------------------
__global__ __launch_bounds__(256)
void decouple_kernel(const float* __restrict__ att,
                     const float* __restrict__ cidp,
                     float* __restrict__ out,
                     float* __restrict__ part_sum,
                     float* __restrict__ part_sq)
{
    __shared__ float red[256];
    const int s = blockIdx.x, t = threadIdx.x;
    const float* row = att + (size_t)s * C_DIM;

    float dot = 0.f;
    for (int c = t; c < C_DIM; c += 256) dot += row[c] * cidp[c];
    red[t] = dot; __syncthreads();
    for (int o = 128; o > 0; o >>= 1) { if (t < o) red[t] += red[t + o]; __syncthreads(); }
    const float w = 1.f / (1.f + expf(-red[0]));
    __syncthreads();

    float psum = 0.f, psq = 0.f;
    for (int c = t; c < C_DIM; c += 256) {
        float af = row[c] - w * cidp[c];
        out[(size_t)c * S_LEN + s] = af;
        psum += af; psq += af * af;
    }
    red[t] = psum; __syncthreads();
    for (int o = 128; o > 0; o >>= 1) { if (t < o) red[t] += red[t + o]; __syncthreads(); }
    if (t == 0) part_sum[s] = red[0];
    __syncthreads();
    red[t] = psq; __syncthreads();
    for (int o = 128; o > 0; o >>= 1) { if (t < o) red[t] += red[t + o]; __syncthreads(); }
    if (t == 0) part_sq[s] = red[0];
}

__global__ __launch_bounds__(256)
void reduce_stats_kernel(const float* __restrict__ part_sum,
                         const float* __restrict__ part_sq,
                         float* __restrict__ stats)
{
    __shared__ float r1[256], r2[256];
    const int t = threadIdx.x;
    float a = 0.f, b = 0.f;
    for (int i = t; i < S_LEN; i += 256) { a += part_sum[i]; b += part_sq[i]; }
    r1[t] = a; r2[t] = b; __syncthreads();
    for (int o = 128; o > 0; o >>= 1) {
        if (t < o) { r1[t] += r1[t + o]; r2[t] += r2[t + o]; }
        __syncthreads();
    }
    if (t == 0) { stats[0] = r1[0]; stats[1] = r2[0]; }
}

__global__ void norm_kernel(float* __restrict__ out, const float* __restrict__ stats, int n)
{
    int i = blockIdx.x * blockDim.x + threadIdx.x;
    if (i >= n) return;
    const float mean = stats[0] / n;
    const float var  = stats[1] / n - mean * mean;
    const float r    = rsqrtf(var + EPS);
    out[i] = (out[i] - mean) * r;
}

// ---------------------------------------------------------------------------
// Host-side orchestration
// ---------------------------------------------------------------------------
extern "C" void kernel_launch(void* const* d_in, const int* in_sizes, int n_in,
                              void* d_out, int out_size, void* d_ws, size_t ws_size,
                              hipStream_t stream)
{
    (void)in_sizes; (void)n_in; (void)out_size; (void)ws_size;

    const float* z      = (const float*)d_in[0];
    const float* cid    = (const float*)d_in[1];   // (2,768) -> flat 1536
    const float* mlp_w  = (const float*)d_in[2];
    const float* mlp_b  = (const float*)d_in[3];
    const float* proj_w = (const float*)d_in[4];
    const float* proj_b = (const float*)d_in[5];
    const float* wq_w   = (const float*)d_in[6];
    const float* wq_b   = (const float*)d_in[7];
    const float* wk_w   = (const float*)d_in[8];
    const float* wk_b   = (const float*)d_in[9];
    const float* wv_w   = (const float*)d_in[10];
    const float* wv_b   = (const float*)d_in[11];
    const float* wo_w   = (const float*)d_in[12];
    const float* wo_b   = (const float*)d_in[13];
    float* out = (float*)d_out;

    // ---- workspace bump allocator (256B aligned) ----
    char* p = (char*)d_ws;
    auto alloc = [&](size_t bytes) -> char* {
        char* r = p; p += (bytes + 255) & ~(size_t)255; return r;
    };
    const size_t WSZ = (size_t)HID * C_DIM;      // 163840 weight elems (q/k/v/o)
    __hip_bfloat16* wq_bf = (__hip_bfloat16*)alloc(WSZ * 2);
    __hip_bfloat16* wk_bf = (__hip_bfloat16*)alloc(WSZ * 2);
    __hip_bfloat16* wv_bf = (__hip_bfloat16*)alloc(WSZ * 2);
    __hip_bfloat16* wo_bf = (__hip_bfloat16*)alloc(WSZ * 2);
    __hip_bfloat16* zfb   = (__hip_bfloat16*)alloc((size_t)S_LEN * C_DIM * 2);
    __hip_bfloat16* Qb    = (__hip_bfloat16*)alloc((size_t)S_LEN * HID * 2);
    __hip_bfloat16* Kb    = (__hip_bfloat16*)alloc((size_t)S_LEN * HID * 2);
    __hip_bfloat16* Vtb   = (__hip_bfloat16*)alloc((size_t)HID * S_LEN * 2); // V^T (HID,S)
    __hip_bfloat16* Pb    = (__hip_bfloat16*)alloc((size_t)S_LEN * S_LEN * 2);
    __hip_bfloat16* Ob    = (__hip_bfloat16*)alloc((size_t)S_LEN * HID * 2);
    float* att    = (float*)alloc((size_t)S_LEN * C_DIM * 4);
    float* iv     = (float*)alloc(HID * 4);
    float* cidp   = (float*)alloc(C_DIM * 4);
    float* maskv  = (float*)alloc(S_LEN * 4);
    float* psum   = (float*)alloc(S_LEN * 4);
    float* psq    = (float*)alloc(S_LEN * 4);
    float* stats  = (float*)alloc(2 * 4);

    // 1) weights -> bf16
    {
        dim3 g((WSZ + 255) / 256), b(256);
        f32_to_bf16_kernel<<<g, b, 0, stream>>>(wq_w, wq_bf, (int)WSZ);
        f32_to_bf16_kernel<<<g, b, 0, stream>>>(wk_w, wk_bf, (int)WSZ);
        f32_to_bf16_kernel<<<g, b, 0, stream>>>(wv_w, wv_bf, (int)WSZ);
        f32_to_bf16_kernel<<<g, b, 0, stream>>>(wo_w, wo_bf, (int)WSZ);
    }
    // 2) LayerNorm + transpose: (C,S) fp32 -> (S,C) bf16
    ln_rows_kernel<<<dim3(S_LEN / 32), dim3(32, 8), 0, stream>>>(z, zfb);
    // 3) tiny identity projections
    small_linear_kernel<<<dim3((HID + C_DIM + 255) / 256), dim3(256), 0, stream>>>(
        cid, mlp_w, mlp_b, proj_w, proj_b, iv, cidp);
    // 4) Q/K/V projections (WMMA): (4096x1280) x (128x1280)^T
    {
        dim3 g(S_LEN / 16, HID / 64), b(32);
        gemm_bf16_wmma<<<g, b, 0, stream>>>(zfb, wq_bf, wq_b, nullptr, Qb, nullptr,
            S_LEN, HID, C_DIM, C_DIM, C_DIM, HID, 1.0f, 0);
        gemm_bf16_wmma<<<g, b, 0, stream>>>(zfb, wk_bf, wk_b, nullptr, Kb, nullptr,
            S_LEN, HID, C_DIM, C_DIM, C_DIM, HID, 1.0f, 0);
        // V stored transposed (HID, S) so it can be the B-operand of A@V
        gemm_bf16_wmma<<<g, b, 0, stream>>>(zfb, wv_bf, wv_b, nullptr, Vtb, nullptr,
            S_LEN, HID, C_DIM, C_DIM, C_DIM, S_LEN, 1.0f, 1);
    }
    // 5) key mask = sigmoid(iv . K_k)
    mask_kernel<<<dim3(S_LEN / 256), dim3(256), 0, stream>>>(Kb, iv, maskv);
    // 6) logits = (Q K^T / sqrt(S)) * mask  (WMMA, fused scale+mask epilogue)
    gemm_bf16_wmma<<<dim3(S_LEN / 16, S_LEN / 64), dim3(32), 0, stream>>>(
        Qb, Kb, nullptr, maskv, Pb, nullptr,
        S_LEN, S_LEN, HID, HID, HID, S_LEN, 1.0f / 64.0f, 0);
    // 7) softmax rows in place (bf16)
    softmax_kernel<<<dim3(S_LEN), dim3(256), 0, stream>>>(Pb);
    // 8) O = P @ V   (B-operand = V^T as (HID rows, S cols))
    gemm_bf16_wmma<<<dim3(S_LEN / 16, HID / 64), dim3(32), 0, stream>>>(
        Pb, Vtb, nullptr, nullptr, Ob, nullptr,
        S_LEN, HID, S_LEN, S_LEN, S_LEN, HID, 1.0f, 0);
    // 9) attended = O @ wo^T + wo_b  (fp32 out)
    gemm_bf16_wmma<<<dim3(S_LEN / 16, C_DIM / 64), dim3(32), 0, stream>>>(
        Ob, wo_bf, wo_b, nullptr, nullptr, att,
        S_LEN, C_DIM, HID, HID, HID, C_DIM, 1.0f, 0);
    // 10) decoupling + transposed store + partial stats
    decouple_kernel<<<dim3(S_LEN), dim3(256), 0, stream>>>(att, cidp, out, psum, psq);
    // 11) deterministic global stats, then joint layer-norm in place
    reduce_stats_kernel<<<dim3(1), dim3(256), 0, stream>>>(psum, psq, stats);
    const int n = S_LEN * C_DIM;
    norm_kernel<<<dim3((n + 255) / 256), dim3(256), 0, stream>>>(out, stats, n);
}
